// Encoder_89404039233721
// MI455X (gfx1250) — matrix operand order repaired and checked by reading, not compiled, not run on hardware
//
#include <hip/hip_runtime.h>
#include <hip/hip_bf16.h>
#include <stddef.h>

// ---------------------------------------------------------------------------
// GRU encoder for MI455X (gfx1250): bf16 WMMA GEMMs with f32 accumulation.
//   B=64, T=100, E=512, U=512.
//   * Embedding gather -> time-major bf16 X [T,B,E].
//   * Per layer: transpose+cast weights to bf16 W^T; precompute time-parallel
//     input projections Gx/Cx as big WMMA GEMMs (M = T*B = 6400).
//   * Sequential recurrence: per step, gates kernel then update kernel,
//     both WMMA with fused sigmoid/tanh epilogues.
//   * GEMM core is templated on K and software-pipelined (next K-step's
//     fragments loaded before current step's WMMAs) to overlap L2 fetch
//     with matrix math instead of s_wait_loadcnt 0 before every v_wmma.
// ---------------------------------------------------------------------------

typedef __bf16 bf16_t;
typedef __attribute__((ext_vector_type(16))) __bf16 v16bf;
typedef __attribute__((ext_vector_type(8)))  __bf16 v8bf;
typedef __attribute__((ext_vector_type(8)))  float  v8f;

#define TT 100
#define BB 64
#define EE 512
#define UU 512

__device__ __forceinline__ float sigmoidf_(float x) {
    return 1.0f / (1.0f + __expf(-x));
}

// A fragment: 16x32 bf16, row-major source [.,K]. Lane L: row m, with
// K-elements e<8 -> k0+8*half+e, e>=8 -> k0+16+8*half+(e-8).
__device__ __forceinline__ v16bf load_afrag(const bf16_t* __restrict__ A, int K,
                                            int m, int k0, int half) {
    const bf16_t* p = A + (size_t)m * K + k0 + 8 * half;
    v8bf lo = *reinterpret_cast<const v8bf*>(p);
    v8bf hi = *reinterpret_cast<const v8bf*>(p + 16);
    return __builtin_shufflevector(lo, hi, 0,1,2,3,4,5,6,7,8,9,10,11,12,13,14,15);
}

// B fragment: 32x16 bf16 from pre-transposed W^T [N,K]. Lane L: col n,
// K-elements e<8 -> k0+8*half+e, e>=8 -> k0+16+8*half+(e-8).
__device__ __forceinline__ v16bf load_bfrag(const bf16_t* __restrict__ Bt, int K,
                                            int n, int k0, int half) {
    const bf16_t* p = Bt + (size_t)n * K + k0 + 8 * half;
    v8bf lo = *reinterpret_cast<const v8bf*>(p);
    v8bf hi = *reinterpret_cast<const v8bf*>(p + 16);
    return __builtin_shufflevector(lo, hi, 0,1,2,3,4,5,6,7,8,9,10,11,12,13,14,15);
}

// Core: one wave computes a 16(M) x 64(N) tile. K known at compile time;
// fully unrolled + software pipelined: iteration k's loads are issued before
// iteration k-1's WMMAs consume their (already resident) fragments.
template <int K>
__device__ __forceinline__ void wmma_16x64(const bf16_t* __restrict__ A,
                                           const bf16_t* __restrict__ Bt,
                                           int m0, int n0, int l16, int half,
                                           v8f (&acc)[4]) {
    const int m = m0 + l16;
    v16bf af = load_afrag(A, K, m, 0, half);
    v16bf b0 = load_bfrag(Bt, K, n0 +  0 + l16, 0, half);
    v16bf b1 = load_bfrag(Bt, K, n0 + 16 + l16, 0, half);
    v16bf b2 = load_bfrag(Bt, K, n0 + 32 + l16, 0, half);
    v16bf b3 = load_bfrag(Bt, K, n0 + 48 + l16, 0, half);
#pragma unroll
    for (int k0 = 0; k0 < K; k0 += 32) {
        v16bf afn, b0n, b1n, b2n, b3n;
        if (k0 + 32 < K) {   // prefetch next K-step before this step's WMMAs
            afn = load_afrag(A, K, m, k0 + 32, half);
            b0n = load_bfrag(Bt, K, n0 +  0 + l16, k0 + 32, half);
            b1n = load_bfrag(Bt, K, n0 + 16 + l16, k0 + 32, half);
            b2n = load_bfrag(Bt, K, n0 + 32 + l16, k0 + 32, half);
            b3n = load_bfrag(Bt, K, n0 + 48 + l16, k0 + 32, half);
        }
        acc[0] = __builtin_amdgcn_wmma_f32_16x16x32_bf16(false, af, false, b0,
                                                         (short)0, acc[0], false, false);
        acc[1] = __builtin_amdgcn_wmma_f32_16x16x32_bf16(false, af, false, b1,
                                                         (short)0, acc[1], false, false);
        acc[2] = __builtin_amdgcn_wmma_f32_16x16x32_bf16(false, af, false, b2,
                                                         (short)0, acc[2], false, false);
        acc[3] = __builtin_amdgcn_wmma_f32_16x16x32_bf16(false, af, false, b3,
                                                         (short)0, acc[3], false, false);
        if (k0 + 32 < K) {
            af = afn; b0 = b0n; b1 = b1n; b2 = b2n; b3 = b3n;
        }
    }
}

// ---------------------------------------------------------------------------
// Kernels
// ---------------------------------------------------------------------------

// Embedding gather -> time-major bf16 [T,B,E].  x is [B,T].
__global__ __launch_bounds__(256) void embed_cast(const int* __restrict__ x,
                                                  const float* __restrict__ emb,
                                                  bf16_t* __restrict__ Xbf) {
    size_t idx = (size_t)blockIdx.x * 256 + threadIdx.x;   // over T*B*E
    int e = (int)(idx % EE);
    int b = (int)((idx / EE) % BB);
    int t = (int)(idx / ((size_t)EE * BB));
    int tok = x[b * TT + t];
    Xbf[idx] = (bf16_t)emb[(size_t)tok * EE + e];
}

// Wt[n*K + k] = (bf16) W[(rowoff + k)*N + n]   (transpose + cast)
__global__ __launch_bounds__(256) void transpose_cast(const float* __restrict__ W,
                                                      bf16_t* __restrict__ Wt,
                                                      int rowoff, int K, int N) {
    size_t idx = (size_t)blockIdx.x * 256 + threadIdx.x;
    if (idx >= (size_t)K * N) return;
    int k = (int)(idx % K);
    int n = (int)(idx / K);
    Wt[idx] = (bf16_t)W[(size_t)(rowoff + k) * N + n];
}

__global__ __launch_bounds__(256) void zero_h(float* __restrict__ h,
                                              bf16_t* __restrict__ hbf) {
    int i = blockIdx.x * 256 + threadIdx.x;   // BB*UU threads
    h[i] = 0.0f;
    hbf[i] = (bf16_t)0.0f;
}

// Out[M,N] = A[M,K](bf16) @ Bt[N,K]^T (bf16) + bias[N], f32 out.
// grid: (M/16)*(N/64)/8 blocks of 256 threads.
template <int K>
__global__ __launch_bounds__(256) void xproj_gemm(const bf16_t* __restrict__ A,
                                                  const bf16_t* __restrict__ Bt,
                                                  const float* __restrict__ bias,
                                                  float* __restrict__ Out,
                                                  int N) {
    const int lane = threadIdx.x & 31;
    const int wv   = threadIdx.x >> 5;
    const int ngrp = N >> 6;
    int gw = blockIdx.x * 8 + wv;
    int m0 = (gw / ngrp) * 16;
    int n0 = (gw % ngrp) * 64;
    int half = lane >> 4, l16 = lane & 15;

    v8f acc[4] = {};
    wmma_16x64<K>(A, Bt, m0, n0, l16, half, acc);

#pragma unroll
    for (int j = 0; j < 4; ++j) {
#pragma unroll
        for (int r = 0; r < 8; ++r) {
            int m = m0 + 8 * half + r;
            int n = n0 + 16 * j + l16;
            Out[(size_t)m * N + n] = acc[j][r] + bias[n];
        }
    }
}

// Gates: ru = sigmoid(Gx[t] + h @ Wgh^T). r-half -> rh_bf = bf16(r*h),
// u-half -> u_scr. M=64, N=1024, K=512. 8 blocks x 256.
__global__ __launch_bounds__(256) void gru_gates(const bf16_t* __restrict__ hbf,
                                                 const bf16_t* __restrict__ WghT,
                                                 const float* __restrict__ Gx_t,
                                                 const float* __restrict__ h,
                                                 float* __restrict__ u_scr,
                                                 bf16_t* __restrict__ rh_bf) {
    const int lane = threadIdx.x & 31;
    const int wv   = threadIdx.x >> 5;
    int gw = blockIdx.x * 8 + wv;          // 64 waves
    int m0 = (gw >> 4) * 16;               // 16 n-groups
    int n0 = (gw & 15) * 64;
    int half = lane >> 4, l16 = lane & 15;

    v8f acc[4] = {};
    wmma_16x64<UU>(hbf, WghT, m0, n0, l16, half, acc);

    // n0 is 64-aligned and the r/u boundary (UU=512) is a multiple of 64,
    // so the whole wave tile lies entirely in one region: uniform branch.
    if (n0 < UU) {
#pragma unroll
        for (int j = 0; j < 4; ++j) {
#pragma unroll
            for (int r = 0; r < 8; ++r) {
                int b = m0 + 8 * half + r;
                int n = n0 + 16 * j + l16;
                float v = sigmoidf_(acc[j][r] + Gx_t[b * (2 * UU) + n]);
                rh_bf[b * UU + n] = (bf16_t)(v * h[b * UU + n]);
            }
        }
    } else {
#pragma unroll
        for (int j = 0; j < 4; ++j) {
#pragma unroll
            for (int r = 0; r < 8; ++r) {
                int b = m0 + 8 * half + r;
                int n = n0 + 16 * j + l16;
                float v = sigmoidf_(acc[j][r] + Gx_t[b * (2 * UU) + n]);
                u_scr[b * UU + (n - UU)] = v;
            }
        }
    }
}

// Update: c = tanh(Cx[t] + rh @ Wch^T); h = u*h + (1-u)*c.
// M=64, N=512, K=512. 4 blocks x 256.
__global__ __launch_bounds__(256) void gru_update(const bf16_t* __restrict__ rh_bf,
                                                  const bf16_t* __restrict__ WchT,
                                                  const float* __restrict__ Cx_t,
                                                  const float* __restrict__ u_scr,
                                                  float* __restrict__ h,
                                                  bf16_t* __restrict__ hbf,
                                                  bf16_t* __restrict__ ybf_t,
                                                  int ystride, int ycol,
                                                  float* __restrict__ yf32_t) {
    const int lane = threadIdx.x & 31;
    const int wv   = threadIdx.x >> 5;
    int gw = blockIdx.x * 8 + wv;          // 32 waves
    int m0 = (gw >> 3) * 16;               // 8 n-groups
    int n0 = (gw & 7) * 64;
    int half = lane >> 4, l16 = lane & 15;

    v8f acc[4] = {};
    wmma_16x64<UU>(rh_bf, WchT, m0, n0, l16, half, acc);

#pragma unroll
    for (int j = 0; j < 4; ++j) {
#pragma unroll
        for (int r = 0; r < 8; ++r) {
            int b = m0 + 8 * half + r;
            int n = n0 + 16 * j + l16;
            float c  = tanhf(acc[j][r] + Cx_t[b * UU + n]);
            float u  = u_scr[b * UU + n];
            float hv = h[b * UU + n];
            float hn = u * hv + (1.0f - u) * c;
            h[b * UU + n]   = hn;
            hbf[b * UU + n] = (bf16_t)hn;
            if (ybf_t)  ybf_t[(size_t)b * ystride + ycol + n] = (bf16_t)hn;
            if (yf32_t) yf32_t[b * UU + n] = hn;
        }
    }
}

// d_out = [output [B,T,U] | s1 [B,U] | s2 [B,U]]
__global__ __launch_bounds__(256) void finalize(const float* __restrict__ y2,
                                                const float* __restrict__ h1,
                                                const float* __restrict__ h2,
                                                float* __restrict__ out) {
    size_t idx = (size_t)blockIdx.x * 256 + threadIdx.x;
    const size_t NOUT = (size_t)BB * TT * UU;
    if (idx < NOUT) {
        int n = (int)(idx % UU);
        int t = (int)((idx / UU) % TT);
        int b = (int)(idx / ((size_t)UU * TT));
        out[idx] = y2[((size_t)t * BB + b) * UU + n];
    } else {
        size_t j = idx - NOUT;
        if (j < (size_t)2 * BB * UU) {
            out[idx] = (j < (size_t)BB * UU) ? h1[j] : h2[j - (size_t)BB * UU];
        }
    }
}

// ---------------------------------------------------------------------------
// Host driver
// ---------------------------------------------------------------------------
extern "C" void kernel_launch(void* const* d_in, const int* in_sizes, int n_in,
                              void* d_out, int out_size, void* d_ws, size_t ws_size,
                              hipStream_t stream) {
    (void)in_sizes; (void)n_in; (void)out_size; (void)ws_size;

    const int*   x   = (const int*)d_in[0];
    const float* emb = (const float*)d_in[1];
    const float* Wg[4]  = {(const float*)d_in[2],  (const float*)d_in[6],
                           (const float*)d_in[10], (const float*)d_in[14]};
    const float* bgv[4] = {(const float*)d_in[3],  (const float*)d_in[7],
                           (const float*)d_in[11], (const float*)d_in[15]};
    const float* Wc[4]  = {(const float*)d_in[4],  (const float*)d_in[8],
                           (const float*)d_in[12], (const float*)d_in[16]};
    const float* bcv[4] = {(const float*)d_in[5],  (const float*)d_in[9],
                           (const float*)d_in[13], (const float*)d_in[17]};
    const int Din[4] = {EE, EE, 2 * UU, UU};   // fw, bw, u1, u2 input dims

    // bump allocator over workspace (256-byte aligned blocks)
    char*  ws  = (char*)d_ws;
    size_t off = 0;
    auto alloc = [&](size_t bytes) -> void* {
        void* p = ws + off;
        off = (off + bytes + 255) & ~(size_t)255;
        return p;
    };

    bf16_t* Xbf = (bf16_t*)alloc((size_t)TT * BB * EE * 2);
    bf16_t *WgxT[4], *WghT[4], *WcxT[4], *WchT[4];
    for (int l = 0; l < 4; ++l) {
        WgxT[l] = (bf16_t*)alloc((size_t)2 * UU * Din[l] * 2);
        WghT[l] = (bf16_t*)alloc((size_t)2 * UU * UU * 2);
        WcxT[l] = (bf16_t*)alloc((size_t)UU * Din[l] * 2);
        WchT[l] = (bf16_t*)alloc((size_t)UU * UU * 2);
    }
    float*  Gx   = (float*) alloc((size_t)TT * BB * 2 * UU * 4);  // reused per layer
    float*  Cx   = (float*) alloc((size_t)TT * BB * UU * 4);      // reused per layer
    bf16_t* ycat = (bf16_t*)alloc((size_t)TT * BB * 2 * UU * 2);  // [yf | yb]
    bf16_t* y1bf = (bf16_t*)alloc((size_t)TT * BB * UU * 2);
    float*  y2   = (float*) alloc((size_t)TT * BB * UU * 4);
    float*  hbuf[4]; bf16_t* hbfb[4];
    for (int l = 0; l < 4; ++l) {
        hbuf[l] = (float*) alloc((size_t)BB * UU * 4);
        hbfb[l] = (bf16_t*)alloc((size_t)BB * UU * 2);
    }
    float*  u_scr = (float*) alloc((size_t)BB * UU * 4);
    bf16_t* rh_bf = (bf16_t*)alloc((size_t)BB * UU * 2);

    // 1) embedding gather + cast
    embed_cast<<<(TT * BB * EE) / 256, 256, 0, stream>>>(x, emb, Xbf);

    // 2) weight transpose + cast (tiny; weights then live in L2)
    auto tlaunch = [&](const float* W, bf16_t* Wt, int rowoff, int K, int N) {
        size_t cnt = (size_t)K * N;
        transpose_cast<<<(unsigned)((cnt + 255) / 256), 256, 0, stream>>>(W, Wt, rowoff, K, N);
    };
    for (int l = 0; l < 4; ++l) {
        tlaunch(Wg[l], WgxT[l], 0,      Din[l], 2 * UU);
        tlaunch(Wg[l], WghT[l], Din[l], UU,     2 * UU);
        tlaunch(Wc[l], WcxT[l], 0,      Din[l], UU);
        tlaunch(Wc[l], WchT[l], Din[l], UU,     UU);
    }

    // 3) per-layer driver
    auto run_layer = [&](int l, const bf16_t* Ain, bool reverse,
                         bf16_t* ybf_base, int ystride, int ycol, float* y32_base) {
        const int M = TT * BB;
        // time-parallel input projections (big WMMA GEMMs)
        if (Din[l] == EE) {
            xproj_gemm<EE><<<(M / 16) * ((2 * UU) / 64) / 8, 256, 0, stream>>>(
                Ain, WgxT[l], bgv[l], Gx, 2 * UU);
            xproj_gemm<EE><<<(M / 16) * (UU / 64) / 8, 256, 0, stream>>>(
                Ain, WcxT[l], bcv[l], Cx, UU);
        } else {   // Din == 2*UU (u1 layer)
            xproj_gemm<2 * UU><<<(M / 16) * ((2 * UU) / 64) / 8, 256, 0, stream>>>(
                Ain, WgxT[l], bgv[l], Gx, 2 * UU);
            xproj_gemm<2 * UU><<<(M / 16) * (UU / 64) / 8, 256, 0, stream>>>(
                Ain, WcxT[l], bcv[l], Cx, UU);
        }
        zero_h<<<(BB * UU) / 256, 256, 0, stream>>>(hbuf[l], hbfb[l]);
        // sequential recurrence
        for (int s = 0; s < TT; ++s) {
            int t = reverse ? (TT - 1 - s) : s;
            gru_gates<<<8, 256, 0, stream>>>(
                hbfb[l], WghT[l], Gx + (size_t)t * BB * 2 * UU,
                hbuf[l], u_scr, rh_bf);
            gru_update<<<4, 256, 0, stream>>>(
                rh_bf, WchT[l], Cx + (size_t)t * BB * UU, u_scr,
                hbuf[l], hbfb[l],
                ybf_base ? ybf_base + (size_t)t * BB * ystride : (bf16_t*)nullptr,
                ystride, ycol,
                y32_base ? y32_base + (size_t)t * BB * UU : (float*)nullptr);
        }
    };

    run_layer(0, Xbf,  false, ycat, 2 * UU, 0,  nullptr);   // fw  -> ycat[:, :U]
    run_layer(1, Xbf,  true,  ycat, 2 * UU, UU, nullptr);   // bw  -> ycat[:, U:]
    run_layer(2, ycat, false, y1bf, UU,     0,  nullptr);   // u1  -> y1bf
    run_layer(3, y1bf, false, nullptr, 0,   0,  y2);        // u2  -> y2

    // 4) output reorder [T,B,U]->[B,T,U] + final states
    size_t total = (size_t)BB * TT * UU + (size_t)2 * BB * UU;
    finalize<<<(unsigned)((total + 255) / 256), 256, 0, stream>>>(
        y2, hbuf[2], hbuf[3], (float*)d_out);
}
